// TimeSeriesPatchEncoder_89335319757135
// MI455X (gfx1250) — compile-verified, matrix operand order checked by reading, not run on hardware
//
#include <hip/hip_runtime.h>

// Problem constants (from reference)
#define BB 32
#define LL 32768
#define MM 32
#define DD 128
#define KK 16
#define SS 8
#define PP 4095            // (LL-KK)/SS + 1
#define CRED (MM*KK)       // 512 reduction length

// Tiling: 8 waves x 32 patches = 256 patches per WG; each wave does two
// 16-row A tiles so every B fragment feeds TWO WMMAs (halves DS-per-WMMA).
#define TILE_P 256
#define NTILES (DD/16)     // 8 n-tiles of 16 columns
#define A_ROWS (TILE_P*SS + SS)   // 2056 x-rows staged per WG
#define W_STRIDE 520              // padded bf16 row stride for W in LDS (1040B, conflict-free)

// LDS carve-up
#define SW_USHORTS (DD * W_STRIDE)                  // 66560 ushorts = 133120 B
#define SA_USHORTS (A_ROWS * MM)                    // 65792 ushorts = 131584 B
#define SMED_OFF_B ((SW_USHORTS + SA_USHORTS) * 2)  // 264704
#define SBIAS_OFF_B (SMED_OFF_B + TILE_P * 4)       // 265728
#define SMEM_BYTES  (SBIAS_OFF_B + DD * 4)          // 266240 (< 320KB WGP LDS)

typedef __attribute__((ext_vector_type(16))) __bf16        v16bf;
typedef __attribute__((ext_vector_type(8)))  float         v8f;
typedef __attribute__((ext_vector_type(4)))  unsigned int  u32x4;
typedef __attribute__((ext_vector_type(8)))  unsigned int  u32x8;

static __device__ __forceinline__ unsigned short f2bf(float f) {
    union { float f; unsigned int u; } v; v.f = f;
    unsigned int u = v.u;
    u += 0x7fffu + ((u >> 16) & 1u);   // round-to-nearest-even (finite inputs)
    return (unsigned short)(u >> 16);
}
static __device__ __forceinline__ unsigned int pack2bf(float lo, float hi) {
    return (unsigned int)f2bf(lo) | ((unsigned int)f2bf(hi) << 16);
}

// Build a 16-element bf16 fragment from two contiguous 16B LDS chunks.
static __device__ __forceinline__ v16bf ld_frag(const unsigned short* b0) {
    u32x4 c0 = *(const u32x4*)b0;            // ds_load_b128
    u32x4 c1 = *(const u32x4*)(b0 + 16);     // ds_load_b128
    u32x8 u;
    u[0] = c0[0]; u[1] = c0[1]; u[2] = c0[2]; u[3] = c0[3];
    u[4] = c1[0]; u[5] = c1[1]; u[6] = c1[2]; u[7] = c1[3];
    return __builtin_bit_cast(v16bf, u);
}

// ---------------------------------------------------------------------------
// Prep: reorder conv_w (D,M,K) fp32 -> bf16 WB[d][c], c = k*32 + m, in d_ws.
// ---------------------------------------------------------------------------
__global__ void tspe_prep_w(const float* __restrict__ w,
                            unsigned short* __restrict__ wb) {
    const int d = blockIdx.x;          // 0..127
    for (int e = threadIdx.x; e < MM * KK; e += blockDim.x) {
        const int m = e >> 4, k = e & 15;
        wb[d * CRED + k * MM + m] = f2bf(w[d * CRED + e]);
    }
}

// ---------------------------------------------------------------------------
// Main: conv-as-GEMM via bf16 WMMA + median + sin/cos PE + bias, fused.
// grid = (ceil(P/256), B), block = 256 (8 waves)
// ---------------------------------------------------------------------------
__global__ void __launch_bounds__(256)
tspe_main(const float* __restrict__ x,
          const float* __restrict__ ts,
          const unsigned short* __restrict__ wb,
          const float* __restrict__ bias,
          float* __restrict__ out) {
    extern __shared__ char smem[];
    unsigned short* sW = (unsigned short*)smem;                 // [DD][W_STRIDE]
    unsigned short* sA = sW + SW_USHORTS;                       // [A_ROWS][MM]
    float* sMed  = (float*)(smem + SMED_OFF_B);                 // [TILE_P]
    float* sBias = (float*)(smem + SBIAS_OFF_B);                // [DD]

    const int b  = blockIdx.y;
    const int p0 = blockIdx.x * TILE_P;
    const int tid = threadIdx.x;

    // ---- Stage W: global bf16 (d_ws) -> LDS with padded stride ----
    for (int c = tid; c < DD * (CRED / 8); c += 256) {          // 16B chunks
        const int d = c >> 6, q = c & 63;
        *(u32x4*)(sW + d * W_STRIDE + q * 8) =
            *(const u32x4*)(wb + d * CRED + q * 8);
    }

    // ---- Stage A: x rows [p0*8, p0*8+A_ROWS) -> bf16 in LDS, zero-fill OOB ----
    {
        const int l0 = p0 * SS;
        int avail = LL - l0; if (avail > A_ROWS) avail = A_ROWS;
        unsigned int* sAd = (unsigned int*)smem + (SW_USHORTS / 2);
        const float* xb = x + ((size_t)b * LL + l0) * MM;       // contiguous window
        for (int idx = tid; idx < A_ROWS * (MM / 2); idx += 256) {
            const int row = idx >> 4;
            unsigned int pk = 0u;
            if (row < avail) {
                const float2 f = ((const float2*)xb)[idx];
                pk = pack2bf(f.x, f.y);
            }
            sAd[idx] = pk;
        }
    }

    // ---- Medians (all 256 threads) + bias (threads 0..127) ----
    {
        const int p = p0 + tid;
        float med = 0.f;
        if (p < PP) {
            const float4* tw = (const float4*)(ts + (size_t)b * LL + p * SS);
            float v[KK];
            #pragma unroll
            for (int q = 0; q < 4; ++q) {                       // 4x float4 window load
                const float4 f = tw[q];
                v[q*4+0] = f.x; v[q*4+1] = f.y; v[q*4+2] = f.z; v[q*4+3] = f.w;
            }
            #pragma unroll
            for (int j = 0; j < KK; ++j) {                      // rank-7 selection
                int rank = 0;
                #pragma unroll
                for (int i = 0; i < KK; ++i)
                    rank += (v[i] < v[j]) || (v[i] == v[j] && i < j);
                if (rank == (KK - 1) / 2) med = v[j];
            }
        }
        sMed[tid] = med;
        if (tid < DD) sBias[tid] = bias[tid];
    }
    __syncthreads();

    // ---- WMMA main loop: each wave -> 32 patches (2 A tiles) x 128 columns ----
    const int wv   = tid >> 5;
    const int lane = tid & 31;
    const int ln   = lane & 15;
    const int kh   = lane >> 4;                                 // K half-select
    const int r0   = wv * 32 + ln;                              // A row, first tile

    v8f acc0[NTILES], acc1[NTILES];
    #pragma unroll
    for (int t = 0; t < NTILES; ++t) {
        v8f z = {0.f, 0.f, 0.f, 0.f, 0.f, 0.f, 0.f, 0.f};
        acc0[t] = z; acc1[t] = z;
    }

    const unsigned short* aBase0 = sA + (r0 * SS) * MM + kh * 8;
    const unsigned short* aBase1 = aBase0 + (16 * SS) * MM;     // +16 patches
    const unsigned short* bBase  = sW + ln * W_STRIDE + kh * 8;

    #pragma unroll
    for (int ks = 0; ks < KK; ++ks) {                           // 16 k-steps of 32
        const v16bf af0 = ld_frag(aBase0 + ks * MM);
        const v16bf af1 = ld_frag(aBase1 + ks * MM);
        #pragma unroll
        for (int t = 0; t < NTILES; ++t) {
            const v16bf bf = ld_frag(bBase + t * (16 * W_STRIDE) + ks * MM);
            acc0[t] = __builtin_amdgcn_wmma_f32_16x16x32_bf16(
                false, af0, false, bf, (short)0, acc0[t], false, false);
            acc1[t] = __builtin_amdgcn_wmma_f32_16x16x32_bf16(
                false, af1, false, bf, (short)0, acc1[t], false, false);
        }
    }

    // ---- Epilogue: + bias + positional encoding, coalesced stores ----
    // C layout: VGPR v, lanes 0-15 -> row v, lanes 16-31 -> row v+8; col = ln.
    // d = t*16 + ln  =>  d parity == ln parity (lane-uniform sin/cos select),
    // div_term/bias depend only on t for a given lane -> hoist.
    float dterm[NTILES], bs[NTILES];
    const bool isCos = (ln & 1);
    #pragma unroll
    for (int t = 0; t < NTILES; ++t) {
        const int d = t * 16 + ln;
        dterm[t] = __expf(-0.14391158f * (float)(d >> 1));      // div_term[d>>1]
        bs[t] = sBias[d];
    }
    #pragma unroll
    for (int s = 0; s < 2; ++s) {
        #pragma unroll
        for (int v = 0; v < 8; ++v) {
            const int row = wv * 32 + s * 16 + v + 8 * kh;
            const int p = p0 + row;
            if (p < PP) {
                const float med = sMed[row];
                float* orow = out + ((size_t)b * PP + p) * DD + ln;
                #pragma unroll
                for (int t = 0; t < NTILES; ++t) {
                    const float ang = med * dterm[t];
                    const float pe = isCos ? __cosf(ang) : __sinf(ang);
                    const float cv = s ? acc1[t][v] : acc0[t][v];
                    orow[t * 16] = cv + bs[t] + pe;
                }
            }
        }
    }
}

// ---------------------------------------------------------------------------
extern "C" void kernel_launch(void* const* d_in, const int* in_sizes, int n_in,
                              void* d_out, int out_size, void* d_ws, size_t ws_size,
                              hipStream_t stream) {
    const float* x    = (const float*)d_in[0];   // (B, L, M) fp32
    const float* ts   = (const float*)d_in[1];   // (B, L)    fp32
    const float* w    = (const float*)d_in[2];   // (D, M, K) fp32
    const float* bias = (const float*)d_in[3];   // (D,)      fp32
    float* out = (float*)d_out;                  // (B, P, D) fp32
    unsigned short* wb = (unsigned short*)d_ws;  // bf16 WB[d][c], 128 KB

    tspe_prep_w<<<dim3(DD), dim3(256), 0, stream>>>(w, wb);

    const int ptiles = (PP + TILE_P - 1) / TILE_P;   // 16
    tspe_main<<<dim3(ptiles, BB), dim3(256), SMEM_BYTES, stream>>>(
        x, ts, wb, bias, out);
}
